// ProtoGraph_59966333387362
// MI455X (gfx1250) — compile-verified, matrix operand order checked by reading, not run on hardware
//
#include <hip/hip_runtime.h>
#include <hip/hip_bf16.h>
#include <math.h>

typedef __attribute__((ext_vector_type(16))) __bf16 v16bf;
typedef __attribute__((ext_vector_type(8)))  float  v8f;

struct alignas(16) U16 { unsigned int w[4]; };
union  FragBF { v16bf v; U16 q[2]; };

#define P_DIM 256
#define D_DIM 512
#define TM    64      // rows per block (4 waves x 16)

// ---- workspace byte offsets ----
#define WS_PB   0                             // proto bf16 [256][512]
#define WS_PTB  (WS_PB  + P_DIM*D_DIM*2)      // protoT bf16 [512][256]
#define WS_MT   (WS_PTB + D_DIM*P_DIM*2)      // M^T bf16 [256][256], M=entmax15(adj)
#define WS_PP   (WS_MT  + P_DIM*P_DIM*2)      // float[256] ||p||^2
#define WS_GATE (WS_PP  + P_DIM*4)            // float[256] gate

// ---- LDS layout (dynamic) ----
#define LDS_PROTO 0                           // bf16 [256][512]  (256 KB)
#define LDS_PANE  (P_DIM*D_DIM*2)             // 4 waves x 16 KB panes
#define LDS_TOTAL (LDS_PANE + 4*16384)        // 327680 B == 320 KB (WGP max)

__device__ __forceinline__ float hsum16(float v) {   // reduce within 16-lane half
  v += __shfl_xor(v, 1, 32); v += __shfl_xor(v, 2, 32);
  v += __shfl_xor(v, 4, 32); v += __shfl_xor(v, 8, 32);
  return v;
}
__device__ __forceinline__ float hmax16(float v) {
  v = fmaxf(v, __shfl_xor(v, 1, 32)); v = fmaxf(v, __shfl_xor(v, 2, 32));
  v = fmaxf(v, __shfl_xor(v, 4, 32)); v = fmaxf(v, __shfl_xor(v, 8, 32));
  return v;
}
__device__ __forceinline__ float wsum32(float v) {
  v = hsum16(v); v += __shfl_xor(v, 16, 32); return v;
}
__device__ __forceinline__ float wmax32(float v) {
  v = hmax16(v); v = fmaxf(v, __shfl_xor(v, 16, 32)); return v;
}

// ---------------- prep: gate = softmax(gate_logits / temp) ----------------
__global__ void k_gate(const float* __restrict__ gl, const float* __restrict__ traw,
                       float* __restrict__ gate) {
  int lane = threadIdx.x;                       // 32 threads
  float temp = 1.f / (1.f + expf(-traw[0])) * (1.f - 0.001f) + 0.001f;
  float v[8]; float m = -3.4e38f;
  #pragma unroll
  for (int i = 0; i < 8; ++i) { v[i] = gl[i * 32 + lane] / temp; m = fmaxf(m, v[i]); }
  m = wmax32(m);
  float s = 0.f;
  #pragma unroll
  for (int i = 0; i < 8; ++i) { v[i] = expf(v[i] - m); s += v[i]; }
  s = wsum32(s);
  #pragma unroll
  for (int i = 0; i < 8; ++i) gate[i * 32 + lane] = v[i] / s;
}

// ------------- prep: M = entmax15(adj_logits) rows; write M^T bf16 -------------
__global__ void k_adj(const float* __restrict__ adj, __bf16* __restrict__ mt) {
  int r = blockIdx.x, lane = threadIdx.x;       // one wave per row
  float z[8]; float m = -3.4e38f;
  #pragma unroll
  for (int i = 0; i < 8; ++i) { z[i] = adj[r * P_DIM + i * 32 + lane] * 0.5f; m = fmaxf(m, z[i]); }
  m = wmax32(m);
  #pragma unroll
  for (int i = 0; i < 8; ++i) z[i] -= m;        // values in (-inf, 0], max==0
  float lo = -1.f, hi = 0.f;
  for (int it = 0; it < 30; ++it) {
    float mid = 0.5f * (lo + hi), s = 0.f;
    #pragma unroll
    for (int i = 0; i < 8; ++i) { float t = fmaxf(z[i] - mid, 0.f); s += t * t; }
    s = wsum32(s);
    if (s >= 1.f) lo = mid; else hi = mid;
  }
  float tau = 0.5f * (lo + hi);
  #pragma unroll
  for (int i = 0; i < 8; ++i) {
    float p = fmaxf(z[i] - tau, 0.f);
    mt[(size_t)(i * 32 + lane) * P_DIM + r] = (__bf16)(p * p);   // transposed store
  }
}

// ------------- prep: bf16 copies of prototypes (+transpose) and ||p||^2 -------------
__global__ void k_proto(const float* __restrict__ proto, __bf16* __restrict__ pb,
                        __bf16* __restrict__ ptb, float* __restrict__ pp) {
  int p = blockIdx.x, tid = threadIdx.x;        // 128 threads, 4 d-values each
  __shared__ float red[4];
  int d0 = tid * 4;
  float4 f = *(const float4*)(proto + (size_t)p * D_DIM + d0);
  float fl[4] = {f.x, f.y, f.z, f.w};
  float s = 0.f;
  #pragma unroll
  for (int k = 0; k < 4; ++k) {
    __bf16 b = (__bf16)fl[k];
    pb[(size_t)p * D_DIM + d0 + k] = b;
    ptb[(size_t)(d0 + k) * P_DIM + p] = b;
    s += fl[k] * fl[k];
  }
  float ws_ = wsum32(s);
  if ((tid & 31) == 0) red[tid >> 5] = ws_;
  __syncthreads();
  if (tid == 0) pp[p] = red[0] + red[1] + red[2] + red[3];
}

// ---------------- fused main kernel ----------------
__global__ void __launch_bounds__(128)
k_main(const float* __restrict__ x, const __bf16* __restrict__ pb,
       const __bf16* __restrict__ mt, const __bf16* __restrict__ ptb,
       const float* __restrict__ pp, const float* __restrict__ gate,
       float* __restrict__ out1, float* __restrict__ out2) {
  extern __shared__ __align__(16) char smem[];
  const int tid = threadIdx.x;
  const int lane16 = tid & 15, cHalf = (tid >> 4) & 1, wave = tid >> 5;
  const int rowBase = blockIdx.x * TM + wave * 16;
  const v8f vzero = {0.f, 0.f, 0.f, 0.f, 0.f, 0.f, 0.f, 0.f};

  // stage prototypes (bf16) into LDS
  {
    const U16* src = (const U16*)pb;
    U16* dst = (U16*)(smem + LDS_PROTO);
    for (int i = tid; i < (P_DIM * D_DIM * 2) / 16; i += 128) dst[i] = src[i];
  }
  __syncthreads();

  char* Sp = smem + LDS_PANE + (size_t)wave * 16384;   // [16][256] bf16: scores, later gated
  char* Wp = Sp + 8192;                                // [16][256] bf16: entmax weights

  // ---- GEMM1: S = x_tile @ proto^T  (bf16 WMMA, K=512, 4 n-groups) + ||x||^2 ----
  float xx = 0.f;
  const float* xr = x + (size_t)(rowBase + lane16) * D_DIM;
  for (int g = 0; g < 4; ++g) {
    v8f a4[4];
    #pragma unroll
    for (int tt = 0; tt < 4; ++tt) a4[tt] = vzero;
    for (int kc = 0; kc < 16; ++kc) {
      const int kA = kc * 32 + cHalf * 8;
      float4 a0 = *(const float4*)(xr + kA);
      float4 a1 = *(const float4*)(xr + kA + 4);
      float4 a2 = *(const float4*)(xr + kA + 16);
      float4 a3 = *(const float4*)(xr + kA + 20);
      FragBF af;
      {
        float fl[16] = {a0.x,a0.y,a0.z,a0.w, a1.x,a1.y,a1.z,a1.w,
                        a2.x,a2.y,a2.z,a2.w, a3.x,a3.y,a3.z,a3.w};
        #pragma unroll
        for (int i = 0; i < 16; ++i) af.v[i] = (__bf16)fl[i];
        if (g == 0) {
          #pragma unroll
          for (int i = 0; i < 16; ++i) xx += fl[i] * fl[i];
        }
      }
      const char* bbase = smem + LDS_PROTO + (size_t)(g * 64 + lane16) * (D_DIM * 2) + (kA << 1);
      #pragma unroll
      for (int tt = 0; tt < 4; ++tt) {
        const char* bp = bbase + (size_t)tt * (16 * D_DIM * 2);
        FragBF bf; bf.q[0] = *(const U16*)bp; bf.q[1] = *(const U16*)(bp + 32);
        a4[tt] = __builtin_amdgcn_wmma_f32_16x16x32_bf16(false, af.v, false, bf.v,
                                                         (short)0, a4[tt], false, false);
      }
    }
    #pragma unroll
    for (int tt = 0; tt < 4; ++tt) {
      #pragma unroll
      for (int j = 0; j < 8; ++j) {
        const int m = cHalf * 8 + j;
        *(__bf16*)(Sp + (size_t)m * 512 + ((g * 4 + tt) * 16 + lane16) * 2) = (__bf16)a4[tt][j];
      }
    }
  }
  xx += __shfl_xor(xx, 16, 32);                 // full ||x_row||^2, row = lane16
  float xxv[8];
  #pragma unroll
  for (int j = 0; j < 8; ++j) xxv[j] = __shfl(xx, cHalf * 8 + j, 32);
  float pv[16];
  #pragma unroll
  for (int t = 0; t < 16; ++t) pv[t] = pp[t * 16 + lane16];

  // ---- entmax15(-dist): read S pane, bisection for tau, write W pane ----
  #pragma unroll 1
  for (int j = 0; j < 8; ++j) {
    const int m = cHalf * 8 + j;
    float zr[16]; float zmax = -3.4e38f;
    #pragma unroll
    for (int t = 0; t < 16; ++t) {
      float sv = (float)(*(const __bf16*)(Sp + (size_t)m * 512 + (t * 16 + lane16) * 2));
      float sq = fmaxf(xxv[j] + pv[t] - 2.f * sv, 0.f);
      float z = -0.5f * sqrtf(sq);              // (-dist)/2
      zr[t] = z; zmax = fmaxf(zmax, z);
    }
    zmax = hmax16(zmax);
    #pragma unroll
    for (int t = 0; t < 16; ++t) zr[t] -= zmax;
    float lo = -1.f, hi = 0.f;
    for (int it = 0; it < 30; ++it) {
      float mid = 0.5f * (lo + hi), s = 0.f;
      #pragma unroll
      for (int t = 0; t < 16; ++t) { float u = fmaxf(zr[t] - mid, 0.f); s += u * u; }
      s = hsum16(s);
      lo = (s >= 1.f) ? mid : lo; hi = (s >= 1.f) ? hi : mid;
    }
    float tau = 0.5f * (lo + hi);
    #pragma unroll
    for (int t = 0; t < 16; ++t) {
      float u = fmaxf(zr[t] - tau, 0.f);
      *(__bf16*)(Wp + (size_t)m * 512 + (t * 16 + lane16) * 2) = (__bf16)(u * u);
    }
  }

  // ---- GEMM2: gated_unnorm = (W @ M) * gate; G pane overwrites S; row sums ----
  float srow[8];
  #pragma unroll
  for (int j = 0; j < 8; ++j) srow[j] = 0.f;
  for (int g = 0; g < 4; ++g) {
    v8f a4[4];
    #pragma unroll
    for (int tt = 0; tt < 4; ++tt) a4[tt] = vzero;
    for (int kc = 0; kc < 8; ++kc) {
      const int kA = kc * 32 + cHalf * 8;
      const char* ap = Wp + (size_t)lane16 * 512 + (kA << 1);
      FragBF af; af.q[0] = *(const U16*)ap; af.q[1] = *(const U16*)(ap + 32);
      const char* bbase = (const char*)mt + ((size_t)(g * 64 + lane16) * P_DIM + kA) * 2;
      __builtin_prefetch(bbase + 64, 0, 1);
      #pragma unroll
      for (int tt = 0; tt < 4; ++tt) {
        const char* bp = bbase + (size_t)tt * (16 * P_DIM * 2);
        FragBF bf; bf.q[0] = *(const U16*)bp; bf.q[1] = *(const U16*)(bp + 32);
        a4[tt] = __builtin_amdgcn_wmma_f32_16x16x32_bf16(false, af.v, false, bf.v,
                                                         (short)0, a4[tt], false, false);
      }
    }
    #pragma unroll
    for (int tt = 0; tt < 4; ++tt) {
      const float gv = gate[(g * 4 + tt) * 16 + lane16];
      #pragma unroll
      for (int j = 0; j < 8; ++j) {
        const int m = cHalf * 8 + j;
        float gg = a4[tt][j] * gv;
        srow[j] += gg;
        *(__bf16*)(Sp + (size_t)m * 512 + ((g * 4 + tt) * 16 + lane16) * 2) = (__bf16)gg;
      }
    }
  }
  float inv[8];
  #pragma unroll
  for (int j = 0; j < 8; ++j) inv[j] = 1.f / (hsum16(srow[j]) + 1e-8f);

  // ---- emit gated = G * inv ----
  #pragma unroll 1
  for (int j = 0; j < 8; ++j) {
    const int m = cHalf * 8 + j;
    const size_t row = (size_t)rowBase + m;
    #pragma unroll
    for (int t = 0; t < 16; ++t) {
      float gb = (float)(*(const __bf16*)(Sp + (size_t)m * 512 + (t * 16 + lane16) * 2));
      out2[row * P_DIM + t * 16 + lane16] = gb * inv[j];
    }
  }

  // ---- GEMM3: blended = inv * (G @ prototypes), D=512 in 8 n-groups ----
  for (int h = 0; h < 8; ++h) {
    v8f a4[4];
    #pragma unroll
    for (int tt = 0; tt < 4; ++tt) a4[tt] = vzero;
    for (int kc = 0; kc < 8; ++kc) {
      const int kA = kc * 32 + cHalf * 8;
      const char* ap = Sp + (size_t)lane16 * 512 + (kA << 1);
      FragBF af; af.q[0] = *(const U16*)ap; af.q[1] = *(const U16*)(ap + 32);
      const char* bbase = (const char*)ptb + ((size_t)(h * 64 + lane16) * P_DIM + kA) * 2;
      __builtin_prefetch(bbase + 64, 0, 1);
      #pragma unroll
      for (int tt = 0; tt < 4; ++tt) {
        const char* bp = bbase + (size_t)tt * (16 * P_DIM * 2);
        FragBF bf; bf.q[0] = *(const U16*)bp; bf.q[1] = *(const U16*)(bp + 32);
        a4[tt] = __builtin_amdgcn_wmma_f32_16x16x32_bf16(false, af.v, false, bf.v,
                                                         (short)0, a4[tt], false, false);
      }
    }
    #pragma unroll
    for (int tt = 0; tt < 4; ++tt) {
      #pragma unroll
      for (int j = 0; j < 8; ++j) {
        const size_t row = (size_t)rowBase + cHalf * 8 + j;
        out1[row * D_DIM + h * 64 + tt * 16 + lane16] = a4[tt][j] * inv[j];
      }
    }
  }
}

extern "C" void kernel_launch(void* const* d_in, const int* in_sizes, int n_in,
                              void* d_out, int out_size, void* d_ws, size_t ws_size,
                              hipStream_t stream) {
  (void)n_in; (void)out_size; (void)ws_size;
  const float* x     = (const float*)d_in[0];
  const float* proto = (const float*)d_in[1];
  const float* adj   = (const float*)d_in[2];
  const float* glog  = (const float*)d_in[3];
  const float* traw  = (const float*)d_in[4];
  const int N = in_sizes[0] / D_DIM;

  char* ws = (char*)d_ws;
  __bf16* pb   = (__bf16*)(ws + WS_PB);
  __bf16* ptb  = (__bf16*)(ws + WS_PTB);
  __bf16* mt   = (__bf16*)(ws + WS_MT);
  float*  pp   = (float*) (ws + WS_PP);
  float*  gate = (float*) (ws + WS_GATE);

  float* out1 = (float*)d_out;                    // blended [N,512]
  float* out2 = out1 + (size_t)N * D_DIM;         // gated   [N,256]

  k_gate <<<1,      32,  0, stream>>>(glog, traw, gate);
  k_adj  <<<P_DIM,  32,  0, stream>>>(adj, mt);
  k_proto<<<P_DIM, 128,  0, stream>>>(proto, pb, ptb, pp);
  k_main <<<N / TM, 128, LDS_TOTAL, stream>>>(x, pb, mt, ptb, pp, gate, out1, out2);
}
